// HMHA_9689446219981
// MI455X (gfx1250) — matrix operand order repaired and checked by reading, not compiled, not run on hardware
//
#include <hip/hip_runtime.h>
#include <hip/hip_bf16.h>
#include <math.h>

// ---------------- CDNA5 (gfx1250) wave32 WMMA helpers ----------------
typedef __bf16 bf16_t;
typedef __attribute__((ext_vector_type(16))) __bf16 v16bf;
typedef __attribute__((ext_vector_type(8)))  __bf16 v8bf;
typedef __attribute__((ext_vector_type(8)))  float  v8f;

#define B_    16
#define CH_   1024
#define HDS_  8
#define D_    32
#define L_    256      // PH*PW
#define TD_   256
#define QKV_  768
#define HW_   64
#define NPIX  4096

__device__ __forceinline__ bf16_t f2bf(float f) {
    union { float f; unsigned u; } in; in.f = f;
    unsigned u = in.u + 0x7FFFu + ((in.u >> 16) & 1u);   // round-to-nearest-even
    union { unsigned short s; bf16_t b; } out; out.s = (unsigned short)(u >> 16);
    return out.b;
}

__device__ __forceinline__ v8f wmma_bf(v16bf a, v16bf b, v8f c) {
    // D = A(16x32 bf16) * B(32x16 bf16) + C(16x16 f32)
    return __builtin_amdgcn_wmma_f32_16x16x32_bf16(false, a, false, b, (short)0, c, false, false);
}

// A fragment: row r per lane (both halves), K = half*8..+7 then 16+half*8..+7
__device__ __forceinline__ v16bf load_a(const bf16_t* rowptr, int k0, int half) {
    union { v16bf v; v8bf h[2]; } a;
    const bf16_t* p = rowptr + k0 + half * 8;
    a.h[0] = *(const v8bf*)(p);
    a.h[1] = *(const v8bf*)(p + 16);
    return a.v;
}
// B fragment: column r per lane, K = half*8..+7 then 16+half*8..+7 (source stored [N][K])
__device__ __forceinline__ v16bf load_b(const bf16_t* colptr, int k0, int half) {
    union { v16bf v; v8bf h[2]; } b;
    const bf16_t* p = colptr + k0 + half * 8;
    b.h[0] = *(const v8bf*)(p);
    b.h[1] = *(const v8bf*)(p + 16);
    return b.v;
}

// ---------------- 1) 4x4 avg-pool + transpose + bf16: x -> xr_t[b][p][c] ----------------
__global__ __launch_bounds__(256) void pool_kernel(const float* __restrict__ x,
                                                   bf16_t* __restrict__ xr_t) {
    int idx = blockIdx.x * 256 + threadIdx.x;       // (b, c, p)
    int p = idx & 255;
    int c = (idx >> 8) & 1023;
    int b = idx >> 18;
    int ph = p >> 4, pw = p & 15;
    const float* src = x + (((size_t)(b * CH_ + c) * HW_) + (size_t)ph * 4) * HW_ + pw * 4;
    float s = 0.f;
#pragma unroll
    for (int r = 0; r < 4; ++r) {
        float4 v = *(const float4*)(src + r * HW_);
        s += v.x + v.y + v.z + v.w;
    }
    xr_t[((size_t)(b * 256 + p)) * CH_ + c] = f2bf(s * (1.f / 16.f));
}

// ---------------- 2) fp32 -> bf16 weight convert ----------------
__global__ __launch_bounds__(256) void convw_kernel(const float* __restrict__ w,
                                                    bf16_t* __restrict__ wb, int n) {
    int i = blockIdx.x * 256 + threadIdx.x;
    if (i < n) wb[i] = f2bf(w[i]);
}

// ---------------- 3) qkv GEMM: qkv[b][o][p] = sum_c Wqkv[o,c]*xr[b,c,p] ----------------
__global__ __launch_bounds__(256) void qkv_gemm_kernel(const bf16_t* __restrict__ wb,
                                                       const bf16_t* __restrict__ xr_t,
                                                       float* __restrict__ qkv) {
    int lane = threadIdx.x & 31, wv = threadIdx.x >> 5;
    int r = lane & 15, half = lane >> 4;
    int o0 = (blockIdx.x * 8 + wv) * 16;            // 48 o-tiles
    int p0 = blockIdx.y * 16;
    int b  = blockIdx.z;
    const bf16_t* aRow = wb   + (size_t)(o0 + r) * CH_;
    const bf16_t* bRow = xr_t + ((size_t)(b * 256) + p0 + r) * CH_;
    v8f acc = {};
    for (int k0 = 0; k0 < CH_; k0 += 32)
        acc = wmma_bf(load_a(aRow, k0, half), load_b(bRow, k0, half), acc);
    float* out = qkv + (size_t)b * QKV_ * L_;
#pragma unroll
    for (int i = 0; i < 8; ++i)
        out[(size_t)(o0 + half * 8 + i) * L_ + p0 + r] = acc[i];
}

// ---------------- 4) fused attention per (b, head): att[b][t][l] ----------------
__global__ __launch_bounds__(256) void attn_kernel(const float* __restrict__ qkv,
                                                   const float* __restrict__ temperature,
                                                   float* __restrict__ att) {
    __shared__ __align__(32) bf16_t q_lds[L_ * D_];   // [l][d] 16KB
    __shared__ __align__(32) bf16_t k_lds[L_ * D_];   // [m][d] 16KB
    __shared__ __align__(32) bf16_t v_lds[D_ * L_];   // [d][m] 16KB
    __shared__ __align__(32) bf16_t p_lds[8 * 16 * L_]; // per-wave 16x256 bf16, 64KB
    int b = blockIdx.x >> 3;
    int h = blockIdx.x & 7;
    int tid = threadIdx.x;
    const float* qb = qkv + (((size_t)b * QKV_) + (size_t)h * D_) * L_;
    const float* kb = qb + (size_t)TD_ * L_;
    const float* vb = qb + (size_t)2 * TD_ * L_;
    {   // normalize q,k into LDS (transposed to [l][d]); v to [d][m]
        int l = tid;
        float tmp[32]; float s = 0.f;
#pragma unroll
        for (int d = 0; d < 32; ++d) { float v = qb[d * L_ + l]; tmp[d] = v; s += v * v; }
        float inv = 1.f / fmaxf(sqrtf(s), 1e-12f);
#pragma unroll
        for (int d = 0; d < 32; ++d) q_lds[l * 32 + d] = f2bf(tmp[d] * inv);
        s = 0.f;
#pragma unroll
        for (int d = 0; d < 32; ++d) { float v = kb[d * L_ + l]; tmp[d] = v; s += v * v; }
        inv = 1.f / fmaxf(sqrtf(s), 1e-12f);
#pragma unroll
        for (int d = 0; d < 32; ++d) k_lds[l * 32 + d] = f2bf(tmp[d] * inv);
        for (int idx = tid; idx < D_ * L_; idx += 256) v_lds[idx] = f2bf(vb[idx]);
    }
    __syncthreads();
    float temp = temperature[h];
    int lane = tid & 31, wv = tid >> 5;
    int r = lane & 15, half = lane >> 4;
    bf16_t* myp = p_lds + wv * (16 * L_);
    for (int lb = wv * 2; lb < wv * 2 + 2; ++lb) {
        v16bf aq = load_a(q_lds + (size_t)(lb * 16 + r) * 32, 0, half);
        v8f acc[16];
#pragma unroll
        for (int mt = 0; mt < 16; ++mt) {
            v16bf bk = load_b(k_lds + (size_t)(mt * 16 + r) * 32, 0, half);
            v8f z = {};
            acc[mt] = wmma_bf(aq, bk, z);
        }
        // softmax over 256 columns (cols spread over 16 lanes x 16 tiles; 8 rows/lane)
        float rm[8], rs[8];
#pragma unroll
        for (int i = 0; i < 8; ++i) rm[i] = -3.0e38f;
#pragma unroll
        for (int mt = 0; mt < 16; ++mt)
#pragma unroll
            for (int i = 0; i < 8; ++i) rm[i] = fmaxf(rm[i], temp * acc[mt][i]);
#pragma unroll
        for (int i = 0; i < 8; ++i) {
#pragma unroll
            for (int m = 1; m < 16; m <<= 1) rm[i] = fmaxf(rm[i], __shfl_xor(rm[i], m, 32));
            rs[i] = 0.f;
        }
#pragma unroll
        for (int mt = 0; mt < 16; ++mt)
#pragma unroll
            for (int i = 0; i < 8; ++i) {
                float e = __expf(temp * acc[mt][i] - rm[i]);
                acc[mt][i] = e; rs[i] += e;
            }
#pragma unroll
        for (int i = 0; i < 8; ++i) {
#pragma unroll
            for (int m = 1; m < 16; m <<= 1) rs[i] += __shfl_xor(rs[i], m, 32);
            rs[i] = 1.f / rs[i];
        }
#pragma unroll
        for (int mt = 0; mt < 16; ++mt)
#pragma unroll
            for (int i = 0; i < 8; ++i)
                myp[(half * 8 + i) * L_ + mt * 16 + r] = f2bf(acc[mt][i] * rs[i]);
        asm volatile("s_wait_dscnt 0" ::: "memory");   // LDS relayout RAW (same wave)
        // attn @ v  (out 16 l x 32 d)
        v8f o0 = {}, o1 = {};
#pragma unroll
        for (int kt = 0; kt < 8; ++kt) {
            v16bf ap = load_a(myp + (size_t)r * L_, kt * 32, half);
            v16bf b0 = load_b(v_lds + (size_t)(0 * 16 + r) * L_, kt * 32, half);
            v16bf b1 = load_b(v_lds + (size_t)(1 * 16 + r) * L_, kt * 32, half);
            o0 = wmma_bf(ap, b0, o0);
            o1 = wmma_bf(ap, b1, o1);
        }
        float* ab = att + ((size_t)b * TD_ + (size_t)h * D_) * L_;
#pragma unroll
        for (int i = 0; i < 8; ++i) {
            int l = lb * 16 + half * 8 + i;
            ab[(size_t)(r) * L_ + l]      = o0[i];   // d = r
            ab[(size_t)(16 + r) * L_ + l] = o1[i];   // d = 16 + r
        }
    }
}

// ---------------- 5) bilinear 16->64 upsample (half-pixel), dual bf16 layouts ----------------
__global__ __launch_bounds__(256) void upsample_kernel(const float* __restrict__ att,
                                                       bf16_t* __restrict__ up_pt,
                                                       bf16_t* __restrict__ up_tp) {
    int idx = blockIdx.x * 256 + threadIdx.x;   // (b, pix)
    int pix = idx & 4095;
    int b = idx >> 12;
    int y = pix >> 6, x = pix & 63;
    float sy = 0.25f * y - 0.375f, sx = 0.25f * x - 0.375f;
    float fy = floorf(sy), fx = floorf(sx);
    float wy1 = sy - fy, wx1 = sx - fx;
    int y0 = (int)fy, x0 = (int)fx;
    int y0c = min(max(y0, 0), 15), y1c = min(max(y0 + 1, 0), 15);
    int x0c = min(max(x0, 0), 15), x1c = min(max(x0 + 1, 0), 15);
    float w00 = (1.f - wy1) * (1.f - wx1), w01 = (1.f - wy1) * wx1;
    float w10 = wy1 * (1.f - wx1),         w11 = wy1 * wx1;
    int q00 = y0c * 16 + x0c, q01 = y0c * 16 + x1c;
    int q10 = y1c * 16 + x0c, q11 = y1c * 16 + x1c;
    const float* a = att + (size_t)b * TD_ * L_;
    bf16_t* op = up_pt + ((size_t)b * NPIX + pix) * TD_;
    bf16_t* ot = up_tp + (size_t)b * TD_ * NPIX + pix;
    for (int t = 0; t < TD_; ++t) {
        const float* at = a + (size_t)t * L_;
        float v = w00 * at[q00] + w01 * at[q01] + w10 * at[q10] + w11 * at[q11];
        bf16_t bv = f2bf(v);
        op[t] = bv;
        ot[(size_t)t * NPIX] = bv;
    }
}

// ---------------- 6) m[t] per batch from att via separable interp weights ----------------
__global__ __launch_bounds__(256) void msum_kernel(const float* __restrict__ att,
                                                   float* __restrict__ m_part) {
    int t = threadIdx.x, b = blockIdx.x;
    float cw[16];
#pragma unroll
    for (int i = 0; i < 16; ++i) cw[i] = 0.f;
    for (int x = 0; x < 64; ++x) {
        float s = 0.25f * x - 0.375f;
        float f = floorf(s);
        float w1 = s - f;
        int i0 = (int)f;
        int i0c = min(max(i0, 0), 15), i1c = min(max(i0 + 1, 0), 15);
        cw[i0c] += 1.f - w1; cw[i1c] += w1;
    }
    const float* a = att + ((size_t)b * TD_ + t) * L_;
    float s = 0.f;
    for (int qy = 0; qy < 16; ++qy) {
        float acc = 0.f;
        for (int qx = 0; qx < 16; ++qx) acc += a[qy * 16 + qx] * cw[qx];
        s += acc * cw[qy];
    }
    m_part[b * 256 + t] = s;
}

// ---------------- 7) S_part[b] = Up_b^T Up_b  (256x256 per batch, WMMA) ----------------
__global__ __launch_bounds__(256) void moment_kernel(const bf16_t* __restrict__ up_tp,
                                                     float* __restrict__ S_part) {
    int lane = threadIdx.x & 31, wv = threadIdx.x >> 5;
    int r = lane & 15, half = lane >> 4;
    int tp0 = (blockIdx.x * 8 + wv) * 16;
    int t0  = blockIdx.y * 16;
    int b   = blockIdx.z;
    const bf16_t* base = up_tp + (size_t)b * TD_ * NPIX;
    const bf16_t* aRow = base + (size_t)(t0  + r) * NPIX;
    const bf16_t* bRow = base + (size_t)(tp0 + r) * NPIX;
    v8f acc = {};
    for (int k0 = 0; k0 < NPIX; k0 += 32)
        acc = wmma_bf(load_a(aRow, k0, half), load_b(bRow, k0, half), acc);
    float* S = S_part + (size_t)b * 65536;
#pragma unroll
    for (int i = 0; i < 8; ++i)
        S[(size_t)(t0 + half * 8 + i) * 256 + tp0 + r] = acc[i];
}

// ---------------- 8) reduce S_part/m_part over batches ----------------
__global__ __launch_bounds__(256) void reduce_kernel(const float* __restrict__ S_part,
                                                     const float* __restrict__ m_part,
                                                     float* __restrict__ S,
                                                     float* __restrict__ m) {
    int i = blockIdx.x * 256 + threadIdx.x;     // 65536
    float s = 0.f;
    for (int b = 0; b < 16; ++b) s += S_part[(size_t)b * 65536 + i];
    S[i] = s;
    if (i < 256) {
        float t = 0.f;
        for (int b = 0; b < 16; ++b) t += m_part[b * 256 + i];
        m[i] = t;
    }
}

// ---------------- 9) fold BN into projection: wf = s_c * W_proj, bias = beta - s_c*mean ----------------
__global__ __launch_bounds__(256) void fold_kernel(const float* __restrict__ S,
                                                   const float* __restrict__ m,
                                                   const float* __restrict__ w_proj,
                                                   const float* __restrict__ gamma,
                                                   const float* __restrict__ beta,
                                                   bf16_t* __restrict__ wf,
                                                   float* __restrict__ bias) {
    __shared__ float red[256];
    __shared__ float wsh[256];
    __shared__ float bc[2];
    int c = blockIdx.x, t = threadIdx.x;
    float w = w_proj[(size_t)c * 256 + t];
    wsh[t] = w;
    __syncthreads();
    float u = 0.f;
    const float* Srow = S + (size_t)t * 256;
    for (int tp = 0; tp < 256; ++tp) u = fmaf(Srow[tp], wsh[tp], u);
    red[t] = w * u;                              // -> sum = w S w^T
    __syncthreads();
    for (int s = 128; s > 0; s >>= 1) { if (t < s) red[t] += red[t + s]; __syncthreads(); }
    float e2sum = red[0];
    __syncthreads();
    red[t] = w * m[t];                           // -> sum = w . m
    __syncthreads();
    for (int s = 128; s > 0; s >>= 1) { if (t < s) red[t] += red[t + s]; __syncthreads(); }
    float msum = red[0];
    if (t == 0) {
        float N = 65536.f;                       // b*h*w
        float mean = msum / N;
        float var = e2sum / N - mean * mean;
        float sc = gamma[c] * rsqrtf(var + 1e-5f);
        bc[0] = sc;
        bc[1] = beta[c] - sc * mean;
        bias[c] = bc[1];
    }
    __syncthreads();
    wf[(size_t)c * 256 + t] = f2bf(w * bc[0]);
}

// ---------------- 10) projection GEMM + bias (write-bound, 268MB out) ----------------
__global__ __launch_bounds__(256) void proj_kernel(const bf16_t* __restrict__ wf,
                                                   const bf16_t* __restrict__ up_pt,
                                                   const float* __restrict__ bias,
                                                   float* __restrict__ out) {
    int lane = threadIdx.x & 31, wv = threadIdx.x >> 5;
    int r = lane & 15, half = lane >> 4;
    int c0  = (blockIdx.y * 8 + wv) * 16;
    int px0 = blockIdx.x * 64;
    int b   = blockIdx.z;
    const bf16_t* aRow  = wf + (size_t)(c0 + r) * TD_;
    const bf16_t* bBase = up_pt + ((size_t)b * NPIX + px0) * TD_;
    v8f a0 = {}, a1 = {}, a2 = {}, a3 = {};
    for (int k0 = 0; k0 < TD_; k0 += 32) {
        v16bf a = load_a(aRow, k0, half);
        v16bf b0 = load_b(bBase + (size_t)(0 * 16 + r) * TD_, k0, half);
        v16bf b1 = load_b(bBase + (size_t)(1 * 16 + r) * TD_, k0, half);
        v16bf b2 = load_b(bBase + (size_t)(2 * 16 + r) * TD_, k0, half);
        v16bf b3 = load_b(bBase + (size_t)(3 * 16 + r) * TD_, k0, half);
        a0 = wmma_bf(a, b0, a0);
        a1 = wmma_bf(a, b1, a1);
        a2 = wmma_bf(a, b2, a2);
        a3 = wmma_bf(a, b3, a3);
    }
    float* ob = out + (size_t)b * CH_ * NPIX;
#pragma unroll
    for (int i = 0; i < 8; ++i) {
        int c = c0 + half * 8 + i;
        float bs = bias[c];
        float* row = ob + (size_t)c * NPIX + px0;
        row[r]      = a0[i] + bs;
        row[16 + r] = a1[i] + bs;
        row[32 + r] = a2[i] + bs;
        row[48 + r] = a3[i] + bs;
    }
}

// ---------------- launch ----------------
extern "C" void kernel_launch(void* const* d_in, const int* in_sizes, int n_in,
                              void* d_out, int out_size, void* d_ws, size_t ws_size,
                              hipStream_t stream) {
    (void)in_sizes; (void)n_in; (void)out_size; (void)ws_size;
    const float* x        = (const float*)d_in[0];
    const float* w_qkv    = (const float*)d_in[1];
    const float* temper   = (const float*)d_in[2];
    const float* w_proj   = (const float*)d_in[3];
    const float* bn_gamma = (const float*)d_in[4];
    const float* bn_beta  = (const float*)d_in[5];
    float* out = (float*)d_out;

    // workspace carve (~94.3 MiB total)
    char* w8 = (char*)d_ws;
    bf16_t* xr_t   = (bf16_t*)(w8);                 //  8,388,608  (b,p,c) bf16
    bf16_t* wqkv_b = (bf16_t*)(w8 + 8388608);       //  1,572,864
    float*  qkv    = (float*) (w8 + 9961472);       // 12,582,912  (b,768,256)
    float*  att    = (float*) (w8 + 22544384);      //  4,194,304  (b,256,256)
    bf16_t* up_pt  = (bf16_t*)(w8 + 26738688);      // 33,554,432  (b,pix,t)
    bf16_t* up_tp  = (bf16_t*)(w8 + 60293120);      // 33,554,432  (b,t,pix)
    float*  S_part = (float*) (w8 + 93847552);      //  4,194,304
    float*  S_sum  = (float*) (w8 + 98041856);      //    262,144
    float*  m_part = (float*) (w8 + 98304000);      //     16,384
    float*  m_sum  = (float*) (w8 + 98320384);      //      1,024
    bf16_t* wf     = (bf16_t*)(w8 + 98321408);      //    524,288
    float*  biasb  = (float*) (w8 + 98845696);      //      4,096

    pool_kernel<<<16384, 256, 0, stream>>>(x, xr_t);
    convw_kernel<<<3072, 256, 0, stream>>>(w_qkv, wqkv_b, QKV_ * CH_);
    qkv_gemm_kernel<<<dim3(6, 16, 16), 256, 0, stream>>>(wqkv_b, xr_t, qkv);
    attn_kernel<<<128, 256, 0, stream>>>(qkv, temper, att);
    upsample_kernel<<<256, 256, 0, stream>>>(att, up_pt, up_tp);
    msum_kernel<<<16, 256, 0, stream>>>(att, m_part);
    moment_kernel<<<dim3(2, 16, 16), 256, 0, stream>>>(up_tp, S_part);
    reduce_kernel<<<256, 256, 0, stream>>>(S_part, m_part, S_sum, m_sum);
    fold_kernel<<<1024, 256, 0, stream>>>(S_sum, m_sum, w_proj, bn_gamma, bn_beta, wf, biasb);
    proj_kernel<<<dim3(64, 8, 16), 256, 0, stream>>>(wf, up_pt, biasb, out);
}
